// PhysicsMaskedRNAModel_86182813762319
// MI455X (gfx1250) — matrix operand order repaired and checked by reading, not compiled, not run on hardware
//
#include <hip/hip_runtime.h>
#include <stdint.h>

// ---------------- problem constants ----------------
#define NN 16384
#define MM 1024
#define HH 256
#define HEADS 8
#define DH 32
#define EE 262144

typedef __bf16 bf16;
typedef __attribute__((ext_vector_type(16))) __bf16 v16bf;
typedef __attribute__((ext_vector_type(8)))  float  v8f;

// gfx1250 async-to-LDS path (probe-guarded; falls back to sync copy)
#if defined(__has_builtin)
#if __has_builtin(__builtin_amdgcn_global_load_async_to_lds_b128)
#define HAVE_ASYNC 1
#endif
#endif
#ifndef HAVE_ASYNC
#define HAVE_ASYNC 0
#endif

#if HAVE_ASYNC
#if __has_builtin(__builtin_amdgcn_s_wait_asynccnt)
#define WAIT_ASYNC(n) __builtin_amdgcn_s_wait_asynccnt(n)
#else
#define WAIT_ASYNC(n) asm volatile("s_wait_asynccnt %0" ::"n"(n) : "memory")
#endif
// builtin expects: (v4i AS(1)*, v4i AS(3)*, imm offset, imm cpol)
typedef int v4i_t __attribute__((vector_size(4 * sizeof(int))));
typedef __attribute__((address_space(1))) v4i_t* gv4_ptr;
typedef __attribute__((address_space(3))) v4i_t* lv4_ptr;
#endif

// ---------------- helpers ----------------
__device__ __forceinline__ unsigned fenc(float f) {
  unsigned u = __float_as_uint(f);
  return (u & 0x80000000u) ? ~u : (u | 0x80000000u);
}
__device__ __forceinline__ float fdec(unsigned u) {
  return __uint_as_float((u & 0x80000000u) ? (u & 0x7FFFFFFFu) : ~u);
}
__device__ __forceinline__ float silu(float x) { return x / (1.0f + __expf(-x)); }

// ---------------- utility kernels ----------------
__global__ void k_zero_f(float* __restrict__ p, int n) {
  int i = blockIdx.x * 256 + threadIdx.x;
  if (i < n) p[i] = 0.0f;
}
__global__ void k_f2bf(const float* __restrict__ in, bf16* __restrict__ out, int n) {
  int i = blockIdx.x * 256 + threadIdx.x;
  if (i < n) out[i] = (bf16)in[i];
}
// convert + transpose: in[K,Ncol] f32 row-major -> out[Ncol,K] bf16 (K-contiguous)
__global__ void k_f2bfT(const float* __restrict__ in, bf16* __restrict__ out,
                        int K, int Ncol) {
  int i = blockIdx.x * 256 + threadIdx.x;
  if (i >= K * Ncol) return;
  int k = i / Ncol, n = i - k * Ncol;
  out[(size_t)n * K + k] = (bf16)in[i];
}

// ---------------- WMMA bf16 GEMM ----------------
// C[Mr,Nc] = A[Mr,K] * B[K,Nc], with B given transposed: Bt[Nc,K].
// Block = 8 waves computing a 64x32 macro-tile (wave (wm,wn) -> 16x16 tile).
// A tile (64x32 bf16) double-buffered in LDS via async-to-LDS when available;
// B fragments are contiguous 32B runs loaded straight from L2.
template <int KD>
__global__ __launch_bounds__(256) void k_gemm(const bf16* __restrict__ A,
                                              const bf16* __restrict__ Bt,
                                              float* __restrict__ C,
                                              int Mr, int Nc) {
  constexpr int NSTEP = KD / 32;
  __shared__ __align__(16) bf16 As[2][64 * 32];
  const int tid = threadIdx.x;
  const int lane = tid & 31, wave = tid >> 5;
  const int wm = wave >> 1, wn = wave & 1;
  const int lmod = lane & 15, lhalf = lane >> 4;
  const int bm = blockIdx.x, bn = blockIdx.y;
  const int srow = tid >> 2, sch = (tid & 3) * 8;  // 16B chunk per thread

  auto stage = [&](int step, int buf) {
    const bf16* gp = A + (size_t)(bm * 64 + srow) * KD + step * 32 + sch;
    bf16* lp = &As[buf][srow * 32 + sch];
#if HAVE_ASYNC
    __builtin_amdgcn_global_load_async_to_lds_b128(
        (gv4_ptr)(uintptr_t)gp, (lv4_ptr)(unsigned)(uintptr_t)lp, 0, 0);
#else
    *(int4*)lp = *(const int4*)gp;
#endif
  };

  stage(0, 0);
  v8f acc = {};
#pragma unroll
  for (int s = 0; s < NSTEP; ++s) {
    const int buf = s & 1;
    if (s + 1 < NSTEP) stage(s + 1, buf ^ 1);
#if HAVE_ASYNC
    if (s + 1 < NSTEP) { WAIT_ASYNC(1); } else { WAIT_ASYNC(0); }
#endif
    __syncthreads();
    v16bf a, b;
    const bf16* Ar = &As[buf][(16 * wm + lmod) * 32];
#pragma unroll
    for (int vv = 0; vv < 8; ++vv) {
      int kb = 2 * vv + lhalf * 16;
      a[2 * vv]     = Ar[kb];
      a[2 * vv + 1] = Ar[kb + 1];
    }
    const bf16* Br =
        Bt + (size_t)(bn * 32 + wn * 16 + lmod) * KD + s * 32 + lhalf * 16;
#pragma unroll
    for (int i = 0; i < 16; ++i) b[i] = Br[i];
    acc = __builtin_amdgcn_wmma_f32_16x16x32_bf16(false, a, false, b, (short)0,
                                                  acc, false, false);
    __syncthreads();
  }
  const int n = bn * 32 + wn * 16 + lmod;
  const int mb = bm * 64 + wm * 16 + lhalf * 8;
#pragma unroll
  for (int r = 0; r < 8; ++r) C[(size_t)(mb + r) * Nc + n] = acc[r];
}

// ---------------- stage 1: atom embedding ----------------
__global__ __launch_bounds__(256) void k_embed(const float* __restrict__ sx,
                                               const float* __restrict__ px,
                                               const float* __restrict__ Ws,
                                               const float* __restrict__ bs,
                                               const float* __restrict__ g1,
                                               const float* __restrict__ b1,
                                               const float* __restrict__ Wp,
                                               float* __restrict__ emb,
                                               bf16* __restrict__ embb) {
  const int row = blockIdx.x, j = threadIdx.x;
  float u = bs[j];
#pragma unroll
  for (int i = 0; i < 4; ++i) u += sx[row * 4 + i] * Ws[i * HH + j];
  __shared__ float sh[HH];
  sh[j] = u; __syncthreads();
  for (int s = 128; s > 0; s >>= 1) { if (j < s) sh[j] += sh[j + s]; __syncthreads(); }
  float mu = sh[0] * (1.0f / HH); __syncthreads();
  float d = u - mu;
  sh[j] = d * d; __syncthreads();
  for (int s = 128; s > 0; s >>= 1) { if (j < s) sh[j] += sh[j + s]; __syncthreads(); }
  float var = sh[0] * (1.0f / HH);
  float y = d * rsqrtf(var + 1e-5f) * g1[j] + b1[j];
  float e = silu(y);
#pragma unroll
  for (int i = 0; i < 10; ++i) e += px[row * 10 + i] * Wp[i * HH + j];
  emb[(size_t)row * HH + j]  = e;
  embb[(size_t)row * HH + j] = (bf16)e;
}

// ---------------- edge attention passes ----------------
__global__ __launch_bounds__(256) void k_scores(const float* __restrict__ q,
                                                const float* __restrict__ k,
                                                const int* __restrict__ ei,
                                                const int* __restrict__ et,
                                                const float* __restrict__ ebias,
                                                float* __restrict__ sc,
                                                unsigned* __restrict__ smax) {
  const int e = blockIdx.x * 8 + (threadIdx.x >> 5);
  const int lane = threadIdx.x & 31;
  if (e >= EE) return;
  const int s = ei[e], d = ei[EE + e], t = et[e];
  const float* qd = q + (size_t)d * HH;
  const float* ks = k + (size_t)s * HH;
#pragma unroll
  for (int h = 0; h < HEADS; ++h) {
    float p = qd[h * DH + lane] * ks[h * DH + lane];
#pragma unroll
    for (int o = 16; o > 0; o >>= 1) p += __shfl_xor(p, o, 32);
    if (lane == 0) {
      float v = p * 0.17677669529663687f + ebias[t * HEADS + h];  // 1/sqrt(32)
      sc[(size_t)e * HEADS + h] = v;
      atomicMax(&smax[(size_t)d * HEADS + h], fenc(v));
    }
  }
}

__global__ void k_exp(const int* __restrict__ ei, float* __restrict__ sc,
                      const unsigned* __restrict__ smax, float* __restrict__ denom) {
  int i = blockIdx.x * 256 + threadIdx.x;
  if (i >= EE * HEADS) return;
  int e = i >> 3, h = i & 7;
  int d = ei[EE + e];
  float ex = __expf(sc[i] - fdec(smax[(size_t)d * HEADS + h]));
  sc[i] = ex;
  atomicAdd(&denom[(size_t)d * HEADS + h], ex);
}

__global__ __launch_bounds__(256) void k_agg(const float* __restrict__ v,
                                             const int* __restrict__ ei,
                                             const float* __restrict__ sc,
                                             const float* __restrict__ denom,
                                             float* __restrict__ agg) {
  const int e = blockIdx.x * 8 + (threadIdx.x >> 5);
  const int lane = threadIdx.x & 31;
  if (e >= EE) return;
  const int s = ei[e], d = ei[EE + e];
#pragma unroll
  for (int h = 0; h < HEADS; ++h) {
    float w = sc[(size_t)e * HEADS + h] / (denom[(size_t)d * HEADS + h] + 1e-9f);
    atomicAdd(&agg[(size_t)d * HH + h * DH + lane],
              w * v[(size_t)s * HH + h * DH + lane]);
  }
}

// ---------------- residual + LN ----------------
__global__ __launch_bounds__(256) void k_ln_res(const float* __restrict__ x,
                                                const float* __restrict__ r,
                                                const float* __restrict__ g,
                                                const float* __restrict__ b,
                                                float* __restrict__ out,
                                                bf16* __restrict__ outb) {
  const int row = blockIdx.x, j = threadIdx.x;
  float u = x[(size_t)row * HH + j] + r[(size_t)row * HH + j];
  __shared__ float sh[HH];
  sh[j] = u; __syncthreads();
  for (int s = 128; s > 0; s >>= 1) { if (j < s) sh[j] += sh[j + s]; __syncthreads(); }
  float mu = sh[0] * (1.0f / HH); __syncthreads();
  float d = u - mu;
  sh[j] = d * d; __syncthreads();
  for (int s = 128; s > 0; s >>= 1) { if (j < s) sh[j] += sh[j + s]; __syncthreads(); }
  float var = sh[0] * (1.0f / HH);
  float y = d * rsqrtf(var + 1e-5f) * g[j] + b[j];
  out[(size_t)row * HH + j]  = y;
  outb[(size_t)row * HH + j] = (bf16)y;
}

// ---------------- pooling logits ----------------
__global__ __launch_bounds__(256) void k_logits(const float* __restrict__ gm,
                                                const float* __restrict__ px,
                                                const float* __restrict__ Wp1,
                                                const float* __restrict__ bp1,
                                                const float* __restrict__ Wp2,
                                                const float* __restrict__ bp2,
                                                const int* __restrict__ anuc,
                                                float* __restrict__ logit,
                                                unsigned* __restrict__ lmax) {
  const int row = blockIdx.x, j = threadIdx.x;
  float u = gm[(size_t)row * HH + j] + bp1[j];
#pragma unroll
  for (int i = 0; i < 10; ++i) u += px[row * 10 + i] * Wp1[i * HH + j];
  float t = tanhf(u);
  __shared__ float sh[HH];
  sh[j] = t * Wp2[j]; __syncthreads();
  for (int s = 128; s > 0; s >>= 1) { if (j < s) sh[j] += sh[j + s]; __syncthreads(); }
  if (j == 0) {
    float L = sh[0] + bp2[0];
    logit[row] = L;
    atomicMax(&lmax[anuc[row]], fenc(L));
  }
}

__global__ void k_lex(float* __restrict__ logit, const unsigned* __restrict__ lmax,
                      const int* __restrict__ anuc, float* __restrict__ lden) {
  int i = blockIdx.x * 256 + threadIdx.x;
  if (i >= NN) return;
  int nm = anuc[i];
  float ex = __expf(logit[i] - fdec(lmax[nm]));
  logit[i] = ex;
  atomicAdd(&lden[nm], ex);
}

__global__ __launch_bounds__(256) void k_pool(const float* __restrict__ logit,
                                              const float* __restrict__ lden,
                                              const int* __restrict__ anuc,
                                              const float* __restrict__ emb2,
                                              float* __restrict__ pooled) {
  const int row = blockIdx.x, j = threadIdx.x;
  int nm = anuc[row];
  float w = logit[row] / (lden[nm] + 1e-9f);
  atomicAdd(&pooled[(size_t)nm * HH + j], w * emb2[(size_t)row * HH + j]);
}

// ---------------- build concat [pooled, h_sugar, h_phos] in bf16 ----------------
__global__ __launch_bounds__(256) void k_cc(const float* __restrict__ pooled,
                                            const float* __restrict__ sug,
                                            const float* __restrict__ pho,
                                            const float* __restrict__ Wsu,
                                            const float* __restrict__ bsu,
                                            const float* __restrict__ Wph,
                                            const float* __restrict__ bph,
                                            bf16* __restrict__ cc) {
  const int m = blockIdx.x, j = threadIdx.x;
  float hs = bsu[j], hp = bph[j];
#pragma unroll
  for (int i = 0; i < 8; ++i) {
    hs += sug[m * 8 + i] * Wsu[i * HH + j];
    hp += pho[m * 8 + i] * Wph[i * HH + j];
  }
  cc[(size_t)m * 768 + j]       = (bf16)pooled[(size_t)m * HH + j];
  cc[(size_t)m * 768 + 256 + j] = (bf16)hs;
  cc[(size_t)m * 768 + 512 + j] = (bf16)hp;
}

// ---------------- nuc = silu(LN(pre + bc)) ----------------
__global__ __launch_bounds__(256) void k_nuc_act(const float* __restrict__ pre,
                                                 const float* __restrict__ bc,
                                                 const float* __restrict__ g,
                                                 const float* __restrict__ b,
                                                 float* __restrict__ out,
                                                 bf16* __restrict__ outb) {
  const int row = blockIdx.x, j = threadIdx.x;
  float u = pre[(size_t)row * HH + j] + bc[j];
  __shared__ float sh[HH];
  sh[j] = u; __syncthreads();
  for (int s = 128; s > 0; s >>= 1) { if (j < s) sh[j] += sh[j + s]; __syncthreads(); }
  float mu = sh[0] * (1.0f / HH); __syncthreads();
  float d = u - mu;
  sh[j] = d * d; __syncthreads();
  for (int s = 128; s > 0; s >>= 1) { if (j < s) sh[j] += sh[j + s]; __syncthreads(); }
  float var = sh[0] * (1.0f / HH);
  float y = silu(d * rsqrtf(var + 1e-5f) * g[j] + b[j]);
  out[(size_t)row * HH + j]  = y;
  outb[(size_t)row * HH + j] = (bf16)y;
}

__global__ void k_bias_silu(float* __restrict__ p, const float* __restrict__ bias, int n) {
  int i = blockIdx.x * 256 + threadIdx.x;
  if (i < n) p[i] = silu(p[i] + bias[i & (HH - 1)]);
}

// ---------------- small heads ----------------
__global__ __launch_bounds__(32) void k_quat(const float* __restrict__ h1,
                                             const float* __restrict__ Wr2,
                                             const float* __restrict__ br2,
                                             float* __restrict__ quat) {
  const int m = blockIdx.x, lane = threadIdx.x;
  float a0 = 0, a1 = 0, a2 = 0, a3 = 0;
  for (int j = lane; j < HH; j += 32) {
    float r = h1[(size_t)m * HH + j];
    a0 += r * Wr2[j * 4 + 0]; a1 += r * Wr2[j * 4 + 1];
    a2 += r * Wr2[j * 4 + 2]; a3 += r * Wr2[j * 4 + 3];
  }
#pragma unroll
  for (int o = 16; o > 0; o >>= 1) {
    a0 += __shfl_xor(a0, o, 32); a1 += __shfl_xor(a1, o, 32);
    a2 += __shfl_xor(a2, o, 32); a3 += __shfl_xor(a3, o, 32);
  }
  if (lane == 0) {
    a0 += br2[0]; a1 += br2[1]; a2 += br2[2]; a3 += br2[3];
    float inv = rsqrtf(a0 * a0 + a1 * a1 + a2 * a2 + a3 * a3);
    quat[m * 4 + 0] = a0 * inv; quat[m * 4 + 1] = a1 * inv;
    quat[m * 4 + 2] = a2 * inv; quat[m * 4 + 3] = a3 * inv;
  }
}

__global__ __launch_bounds__(32) void k_trans(const float* __restrict__ h1,
                                              const float* __restrict__ Wt2,
                                              const float* __restrict__ bt2,
                                              float* __restrict__ tr) {
  const int m = blockIdx.x, lane = threadIdx.x;
  float a0 = 0, a1 = 0, a2 = 0;
  for (int j = lane; j < HH; j += 32) {
    float r = h1[(size_t)m * HH + j];
    a0 += r * Wt2[j * 3 + 0]; a1 += r * Wt2[j * 3 + 1]; a2 += r * Wt2[j * 3 + 2];
  }
#pragma unroll
  for (int o = 16; o > 0; o >>= 1) {
    a0 += __shfl_xor(a0, o, 32); a1 += __shfl_xor(a1, o, 32); a2 += __shfl_xor(a2, o, 32);
  }
  if (lane == 0) {
    tr[m * 3 + 0] = a0 + bt2[0]; tr[m * 3 + 1] = a1 + bt2[1]; tr[m * 3 + 2] = a2 + bt2[2];
  }
}

// ---------------- apply rigid transform per atom ----------------
__global__ void k_coords(const float* __restrict__ quat, const float* __restrict__ tr,
                         const float* __restrict__ coords, const int* __restrict__ anuc,
                         float* __restrict__ out) {
  int i = blockIdx.x * 256 + threadIdx.x;
  if (i >= NN) return;
  int nm = anuc[i];
  float w = quat[nm * 4 + 0], x = quat[nm * 4 + 1];
  float y = quat[nm * 4 + 2], z = quat[nm * 4 + 3];
  float c0 = coords[i * 3 + 0], c1 = coords[i * 3 + 1], c2 = coords[i * 3 + 2];
  float r00 = 1 - 2 * (y * y + z * z), r01 = 2 * (x * y - z * w), r02 = 2 * (x * z + y * w);
  float r10 = 2 * (x * y + z * w), r11 = 1 - 2 * (x * x + z * z), r12 = 2 * (y * z - x * w);
  float r20 = 2 * (x * z - y * w), r21 = 2 * (y * z + x * w), r22 = 1 - 2 * (x * x + y * y);
  out[i * 3 + 0] = r00 * c0 + r01 * c1 + r02 * c2 + tr[nm * 3 + 0];
  out[i * 3 + 1] = r10 * c0 + r11 * c1 + r12 * c2 + tr[nm * 3 + 1];
  out[i * 3 + 2] = r20 * c0 + r21 * c1 + r22 * c2 + tr[nm * 3 + 2];
}

// ---------------- host orchestration ----------------
extern "C" void kernel_launch(void* const* d_in, const int* in_sizes, int n_in,
                              void* d_out, int out_size, void* d_ws, size_t ws_size,
                              hipStream_t stream) {
  (void)in_sizes; (void)n_in; (void)out_size; (void)ws_size;
  const float* physics = (const float*)d_in[0];
  const float* structural = (const float*)d_in[1];
  const float* sugar = (const float*)d_in[2];
  const float* phos = (const float*)d_in[3];
  const float* coords = (const float*)d_in[4];
  const float* Ws = (const float*)d_in[5];
  const float* bs = (const float*)d_in[6];
  const float* ln1g = (const float*)d_in[7];
  const float* ln1b = (const float*)d_in[8];
  const float* Wp = (const float*)d_in[9];
  const float* Wq = (const float*)d_in[10];
  const float* Wk = (const float*)d_in[11];
  const float* Wv = (const float*)d_in[12];
  const float* Wo = (const float*)d_in[13];
  const float* ebias = (const float*)d_in[14];
  const float* ln2g = (const float*)d_in[15];
  const float* ln2b = (const float*)d_in[16];
  const float* Wsu = (const float*)d_in[17];
  const float* bsu = (const float*)d_in[18];
  const float* Wph = (const float*)d_in[19];
  const float* bph = (const float*)d_in[20];
  const float* Wp1 = (const float*)d_in[21];
  const float* bp1 = (const float*)d_in[22];
  const float* Wp2 = (const float*)d_in[23];
  const float* bp2 = (const float*)d_in[24];
  const float* Wc = (const float*)d_in[25];
  const float* bc = (const float*)d_in[26];
  const float* ln3g = (const float*)d_in[27];
  const float* ln3b = (const float*)d_in[28];
  const float* Wr1 = (const float*)d_in[29];
  const float* br1 = (const float*)d_in[30];
  const float* Wr2 = (const float*)d_in[31];
  const float* br2 = (const float*)d_in[32];
  const float* Wt1 = (const float*)d_in[33];
  const float* bt1 = (const float*)d_in[34];
  const float* Wt2 = (const float*)d_in[35];
  const float* bt2 = (const float*)d_in[36];
  const int* anuc = (const int*)d_in[37];
  const int* eidx = (const int*)d_in[38];
  const int* etyp = (const int*)d_in[39];

  size_t off = 0;
  char* base = (char*)d_ws;
  auto take = [&](size_t bytes) -> char* {
    char* p = base + off;
    off = (off + bytes + 255) & ~(size_t)255;
    return p;
  };
  float* emb   = (float*)take((size_t)NN * HH * 4);
  bf16*  actbf = (bf16*) take((size_t)NN * HH * 2);   // emb_bf -> agg_bf -> emb2_bf
  float* qbuf  = (float*)take((size_t)NN * HH * 4);   // q, later gemm tmp
  float* kbuf  = (float*)take((size_t)NN * HH * 4);   // k, later emb2 f32
  float* vbuf  = (float*)take((size_t)NN * HH * 4);
  float* sc    = (float*)take((size_t)EE * HEADS * 4);
  unsigned* smax = (unsigned*)take((size_t)NN * HEADS * 4);
  float* denom = (float*)take((size_t)NN * HEADS * 4);
  float* agg   = (float*)take((size_t)NN * HH * 4);   // agg, later Wp1-gemm tmp
  float* logit = (float*)take((size_t)NN * 4);
  unsigned* lmax = (unsigned*)take((size_t)MM * 4);
  float* lden  = (float*)take((size_t)MM * 4);
  float* pooled = (float*)take((size_t)MM * HH * 4);
  bf16*  ccbf  = (bf16*) take((size_t)MM * 768 * 2);
  float* nucpre = (float*)take((size_t)MM * HH * 4);
  float* nucf  = (float*)take((size_t)MM * HH * 4);
  bf16*  nucbf = (bf16*) take((size_t)MM * HH * 2);
  float* hpre  = (float*)take((size_t)MM * HH * 4);
  float* quatb = (float*)take((size_t)MM * 4 * 4);
  float* trb   = (float*)take((size_t)MM * 3 * 4);
  // transposed bf16 weights: Wt[n*K + k]
  bf16* wqb  = (bf16*)take((size_t)HH * HH * 2);
  bf16* wkb  = (bf16*)take((size_t)HH * HH * 2);
  bf16* wvb  = (bf16*)take((size_t)HH * HH * 2);
  bf16* wob  = (bf16*)take((size_t)HH * HH * 2);
  bf16* wp1b = (bf16*)take((size_t)HH * HH * 2);
  bf16* wcb  = (bf16*)take((size_t)768 * HH * 2);
  bf16* wr1b = (bf16*)take((size_t)HH * HH * 2);
  bf16* wt1b = (bf16*)take((size_t)HH * HH * 2);

  const int TPB = 256;
  auto gs = [](int n) { return (n + 255) / 256; };

  // zero accumulators
  k_zero_f<<<gs(NN * HH), TPB, 0, stream>>>(agg, NN * HH);
  k_zero_f<<<gs(NN * HEADS), TPB, 0, stream>>>(denom, NN * HEADS);
  k_zero_f<<<gs(NN * HEADS), TPB, 0, stream>>>((float*)smax, NN * HEADS);
  k_zero_f<<<gs(MM * HH), TPB, 0, stream>>>(pooled, MM * HH);
  k_zero_f<<<gs(MM), TPB, 0, stream>>>(lden, MM);
  k_zero_f<<<gs(MM), TPB, 0, stream>>>((float*)lmax, MM);

  // weight conversion (f32 -> transposed bf16)
  k_f2bfT<<<gs(HH * HH), TPB, 0, stream>>>(Wq, wqb, HH, HH);
  k_f2bfT<<<gs(HH * HH), TPB, 0, stream>>>(Wk, wkb, HH, HH);
  k_f2bfT<<<gs(HH * HH), TPB, 0, stream>>>(Wv, wvb, HH, HH);
  k_f2bfT<<<gs(HH * HH), TPB, 0, stream>>>(Wo, wob, HH, HH);
  k_f2bfT<<<gs(HH * HH), TPB, 0, stream>>>(Wp1 + 10 * HH, wp1b, HH, HH);
  k_f2bfT<<<gs(768 * HH), TPB, 0, stream>>>(Wc, wcb, 768, HH);
  k_f2bfT<<<gs(HH * HH), TPB, 0, stream>>>(Wr1, wr1b, HH, HH);
  k_f2bfT<<<gs(HH * HH), TPB, 0, stream>>>(Wt1, wt1b, HH, HH);

  // stage 1: atom embedding
  k_embed<<<NN, TPB, 0, stream>>>(structural, physics, Ws, bs, ln1g, ln1b, Wp, emb, actbf);

  // q/k/v projections (WMMA, 64x32 macro-tiles)
  dim3 gN(NN / 64, HH / 32);
  k_gemm<HH><<<gN, TPB, 0, stream>>>(actbf, wqb, qbuf, NN, HH);
  k_gemm<HH><<<gN, TPB, 0, stream>>>(actbf, wkb, kbuf, NN, HH);
  k_gemm<HH><<<gN, TPB, 0, stream>>>(actbf, wvb, vbuf, NN, HH);

  // edge attention: scores -> softmax -> aggregate
  k_scores<<<EE / 8, TPB, 0, stream>>>(qbuf, kbuf, eidx, etyp, ebias, sc, smax);
  k_exp<<<gs(EE * HEADS), TPB, 0, stream>>>(eidx, sc, smax, denom);
  k_agg<<<EE / 8, TPB, 0, stream>>>(vbuf, eidx, sc, denom, agg);

  // agg @ Wo, residual + LN2
  k_f2bf<<<gs(NN * HH), TPB, 0, stream>>>(agg, actbf, NN * HH);
  k_gemm<HH><<<gN, TPB, 0, stream>>>(actbf, wob, qbuf, NN, HH);
  k_ln_res<<<NN, TPB, 0, stream>>>(emb, qbuf, ln2g, ln2b, kbuf, actbf);  // emb2 in kbuf

  // pooling logits: emb2 @ Wp1[10:] (WMMA) + physics part, tanh, @ Wp2
  k_gemm<HH><<<gN, TPB, 0, stream>>>(actbf, wp1b, agg, NN, HH);
  k_logits<<<NN, TPB, 0, stream>>>(agg, physics, Wp1, bp1, Wp2, bp2, anuc, logit, lmax);
  k_lex<<<gs(NN), TPB, 0, stream>>>(logit, lmax, anuc, lden);
  k_pool<<<NN, TPB, 0, stream>>>(logit, lden, anuc, kbuf, pooled);

  // build concat, nuc GEMM + LN3 + silu
  k_cc<<<MM, TPB, 0, stream>>>(pooled, sugar, phos, Wsu, bsu, Wph, bph, ccbf);
  dim3 gM(MM / 64, HH / 32);
  k_gemm<768><<<gM, TPB, 0, stream>>>(ccbf, wcb, nucpre, MM, HH);
  k_nuc_act<<<MM, TPB, 0, stream>>>(nucpre, bc, ln3g, ln3b, nucf, nucbf);

  // quaternion head
  k_gemm<HH><<<gM, TPB, 0, stream>>>(nucbf, wr1b, hpre, MM, HH);
  k_bias_silu<<<gs(MM * HH), TPB, 0, stream>>>(hpre, br1, MM * HH);
  k_quat<<<MM, 32, 0, stream>>>(hpre, Wr2, br2, quatb);

  // translation head
  k_gemm<HH><<<gM, TPB, 0, stream>>>(nucbf, wt1b, hpre, MM, HH);
  k_bias_silu<<<gs(MM * HH), TPB, 0, stream>>>(hpre, bt1, MM * HH);
  k_trans<<<MM, 32, 0, stream>>>(hpre, Wt2, bt2, trb);

  // final rigid transform
  k_coords<<<gs(NN), TPB, 0, stream>>>(quatb, trb, coords, anuc, (float*)d_out);
}
